// MultiheadAttention_463856468444
// MI455X (gfx1250) — matrix operand order repaired and checked by reading, not compiled, not run on hardware
//
#include <hip/hip_runtime.h>
#include <hip/hip_bf16.h>

typedef __attribute__((ext_vector_type(16))) __bf16 v16bf;
typedef __attribute__((ext_vector_type(8)))  __bf16 v8bf;
typedef __attribute__((ext_vector_type(8)))  float  v8f;

#define D_MODEL 1024
#define S_LEN   2048
#define NH      16
#define HD      64
#define BATCH   2
#define ROWS    (BATCH * S_LEN)   /* 4096 */

// ---------------------------------------------------------------------------
// helpers
// ---------------------------------------------------------------------------
static __device__ __forceinline__ v16bf frag_cat(const __bf16* p0, const __bf16* p1) {
  v8bf lo = *(const v8bf*)p0;
  v8bf hi = *(const v8bf*)p1;
  return __builtin_shufflevector(lo, hi, 0,1,2,3,4,5,6,7,8,9,10,11,12,13,14,15);
}

static __device__ __forceinline__ v8f wmma_bf16(v16bf a, v16bf b, v8f c) {
  // D = A(16x32 bf16) * B(32x16 bf16) + C(16x16 f32)
  return __builtin_amdgcn_wmma_f32_16x16x32_bf16(false, a, false, b, (short)0, c, false, false);
}

// ---------------------------------------------------------------------------
// K0: fp32 -> bf16 conversion
// ---------------------------------------------------------------------------
__global__ __launch_bounds__(256) void cvt_f32_bf16(const float* __restrict__ src,
                                                    __bf16* __restrict__ dst, int n) {
  int i = blockIdx.x * blockDim.x + threadIdx.x;
  int stride = gridDim.x * blockDim.x;
  for (; i < n; i += stride) dst[i] = (__bf16)src[i];
}

// ---------------------------------------------------------------------------
// K1: Y[s][n] = sum_m X[s][m] * W[n][m] + bias[n]
// One wave computes a 32(M) x 64(N) macro-tile: 8 WMMAs / 32-wide K step.
// TRANSV=0: scatter to [b,h,s,d].  TRANSV=1: scatter to [b,h,d,s] (V).
// ---------------------------------------------------------------------------
template <int TRANSV>
__global__ __launch_bounds__(256) void proj_gemm_head(const __bf16* __restrict__ X,
                                                      const __bf16* __restrict__ W,
                                                      const float*  __restrict__ bias,
                                                      __bf16* __restrict__ Yh) {
  const int wave   = threadIdx.x >> 5;
  const int lane   = threadIdx.x & 31;
  const int waveId = blockIdx.x * 8 + wave;
  const int mtile  = waveId >> 4;      // 0..127
  const int ntile  = waveId & 15;      // 0..15
  const int laneN  = lane & 15;
  const int hi16   = lane >> 4;
  const int kbA = hi16 * 8;
  const int kbB = hi16 * 16;

  const __bf16* arow0 = X + (size_t)(mtile * 32 + laneN) * D_MODEL;
  const __bf16* arow1 = arow0 + (size_t)16 * D_MODEL;
  const __bf16* brow[4];
#pragma unroll
  for (int t = 0; t < 4; ++t)
    brow[t] = W + (size_t)(ntile * 64 + t * 16 + laneN) * D_MODEL;

  v8f acc[2][4];
#pragma unroll
  for (int mi = 0; mi < 2; ++mi)
#pragma unroll
    for (int t = 0; t < 4; ++t) acc[mi][t] = (v8f){};

#pragma unroll 2
  for (int k = 0; k < D_MODEL; k += 32) {
    v16bf a0 = frag_cat(arow0 + k + kbA, arow0 + k + kbA + 16);
    v16bf a1 = frag_cat(arow1 + k + kbA, arow1 + k + kbA + 16);
#pragma unroll
    for (int t = 0; t < 4; ++t) {
      v16bf b = frag_cat(brow[t] + k + kbA, brow[t] + k + kbA + 16);
      acc[0][t] = wmma_bf16(a0, b, acc[0][t]);
      acc[1][t] = wmma_bf16(a1, b, acc[1][t]);
    }
  }

  const int mbase = hi16 * 8;
  const int h = ntile;                       // 64-col tile == one head
#pragma unroll
  for (int mi = 0; mi < 2; ++mi) {
    const int rowb = mtile * 32 + mi * 16 + mbase;   // 8 consecutive rows
    const int bidx = rowb >> 11;
    const int sb   = rowb & 2047;
#pragma unroll
    for (int t = 0; t < 4; ++t) {
      const int dh = t * 16 + laneN;
      const float bv = bias[h * HD + dh];
      if (TRANSV) {
        v8bf o;
#pragma unroll
        for (int r = 0; r < 8; ++r) o[r] = (__bf16)(acc[mi][t][r] + bv);
        *(v8bf*)(Yh + (((size_t)(bidx * NH + h)) * HD + dh) * S_LEN + sb) = o;
      } else {
#pragma unroll
        for (int r = 0; r < 8; ++r)
          Yh[(((size_t)(bidx * NH + h)) * S_LEN + (sb + r)) * HD + dh] =
              (__bf16)(acc[mi][t][r] + bv);
      }
    }
  }
}

// ---------------------------------------------------------------------------
// K2: flash attention, fully transposed pipeline. One wave per (b,h,16-q tile).
//   scoresT = K(A) x Q(B)  -> lane owns one q column: scalar online softmax
//   O^T    += V^T(A) x P^T(B), P^T built from exp values with 8 shfl_xor(16)
// Q,K in [b,h,s,d]; V transposed in [b,h,d,s]; no LDS needed.
// ---------------------------------------------------------------------------
__global__ __launch_bounds__(256) void attn_flash(const __bf16* __restrict__ Qh,
                                                  const __bf16* __restrict__ Kh,
                                                  const __bf16* __restrict__ Vt,
                                                  __bf16* __restrict__ ctx) {
  const int wave   = threadIdx.x >> 5;
  const int lane   = threadIdx.x & 31;
  const int waveId = blockIdx.x * 8 + wave;
  const int bh = waveId >> 7;           // 0..31
  const int qt = waveId & 127;          // 0..127
  const int laneN = lane & 15;
  const int hi16  = lane >> 4;
  const int mbase = hi16 * 8;

  const float SCALE = 1.0f / 32.0f;     // 1/sqrt(D_MODEL), faithful quirk
  const size_t base = (size_t)bh * S_LEN * HD;

  // Q as B-fragments: b[i] = Q[q=laneN][dstep*32 + hi16*8 + i (+8 for i>=8)]
  const __bf16* qrow = Qh + base + (size_t)(qt * 16 + laneN) * HD;
  v16bf qb0 = frag_cat(qrow + hi16 * 8,      qrow + hi16 * 8 + 16);
  v16bf qb1 = frag_cat(qrow + 32 + hi16 * 8, qrow + 32 + hi16 * 8 + 16);

  // O^T accumulators: 4 d-tiles (M), q columns (N)
  v8f o0 = {}, o1 = {}, o2 = {}, o3 = {};
  float runm = -1e30f;                  // per-lane: stats for q = laneN
  float runl = 0.0f;

  const __bf16* krow = Kh + base + (size_t)laneN * HD;            // + key*HD
  const __bf16* vrow[4];
#pragma unroll
  for (int t = 0; t < 4; ++t)
    vrow[t] = Vt + base + (size_t)(t * 16 + laneN) * S_LEN;       // + key

  for (int kb = 0; kb < S_LEN; kb += 32) {
    if (kb + 32 < S_LEN) {
      __builtin_prefetch(krow + (size_t)(kb + 32) * HD, 0, 1);
      __builtin_prefetch(vrow[0] + kb + 32, 0, 1);
    }

    // ---- scores^T: two 16-key tiles (M) x 16 q (N), K-dim 64 in 2x32
    const __bf16* k0 = krow + (size_t)kb * HD;          // rows: keys kb+laneN
    const __bf16* k1 = k0 + (size_t)16 * HD;            // rows: keys kb+16+laneN
    v16bf a00 = frag_cat(k0 + hi16 * 8,      k0 + hi16 * 8 + 16);
    v16bf a01 = frag_cat(k0 + 32 + hi16 * 8, k0 + 32 + hi16 * 8 + 16);
    v16bf a10 = frag_cat(k1 + hi16 * 8,      k1 + hi16 * 8 + 16);
    v16bf a11 = frag_cat(k1 + 32 + hi16 * 8, k1 + 32 + hi16 * 8 + 16);

    // V^T A-fragments (independent of softmax; issued early)
    v16bf va0 = frag_cat(vrow[0] + kb + hi16 * 16, vrow[0] + kb + hi16 * 16 + 8);
    v16bf va1 = frag_cat(vrow[1] + kb + hi16 * 16, vrow[1] + kb + hi16 * 16 + 8);
    v16bf va2 = frag_cat(vrow[2] + kb + hi16 * 16, vrow[2] + kb + hi16 * 16 + 8);
    v16bf va3 = frag_cat(vrow[3] + kb + hi16 * 16, vrow[3] + kb + hi16 * 16 + 8);

    v8f c0 = {}, c1 = {};
    c0 = wmma_bf16(a00, qb0, c0);
    c1 = wmma_bf16(a10, qb0, c1);
    c0 = wmma_bf16(a01, qb1, c0);
    c1 = wmma_bf16(a11, qb1, c1);

    // ---- online softmax for q = laneN: in-lane reduce + one half-swap
    float mx = fmaxf(c0[0], c1[0]);
#pragma unroll
    for (int r = 1; r < 8; ++r) mx = fmaxf(mx, fmaxf(c0[r], c1[r]));
    mx = fmaxf(mx, __shfl_xor(mx, 16, 32));
    const float mxs = mx * SCALE;

    const float nm = fmaxf(runm, mxs);
    const float al = __expf(runm - nm);
    runm = nm;

    float e0[8], e1[8];
    float bs = 0.0f;
#pragma unroll
    for (int r = 0; r < 8; ++r) {
      e0[r] = __expf(c0[r] * SCALE - nm);
      e1[r] = __expf(c1[r] * SCALE - nm);
      bs += e0[r] + e1[r];
    }
    bs += __shfl_xor(bs, 16, 32);
    runl = runl * al + bs;

#pragma unroll
    for (int r = 0; r < 8; ++r) { o0[r] *= al; o1[r] *= al; o2[r] *= al; o3[r] *= al; }

    // ---- build P^T B-fragment: lane sends the half its partner needs
    v16bf pb;
#pragma unroll
    for (int i = 0; i < 8; ++i) {
      const float snd = hi16 ? e0[i] : e1[i];
      const float rcv = __shfl_xor(snd, 16, 32);
      pb[i]     = (__bf16)(hi16 ? rcv : e0[i]);   // keys kbB+0..7
      pb[8 + i] = (__bf16)(hi16 ? e1[i] : rcv);   // keys kbB+8..15
    }

    // ---- O^T += V^T * P^T
    o0 = wmma_bf16(va0, pb, o0);
    o1 = wmma_bf16(va1, pb, o1);
    o2 = wmma_bf16(va2, pb, o2);
    o3 = wmma_bf16(va3, pb, o3);
  }

  // ---- normalize; O^T rows are consecutive d -> 16B stores
  const int bidx = bh >> 4;
  const int h    = bh & 15;
  const int qg   = qt * 16 + laneN;
  const float inv = 1.0f / runl;
  __bf16* out = ctx + ((size_t)(bidx * S_LEN + qg)) * D_MODEL + h * HD + mbase;

  v8bf s0, s1, s2, s3;
#pragma unroll
  for (int r = 0; r < 8; ++r) {
    s0[r] = (__bf16)(o0[r] * inv);
    s1[r] = (__bf16)(o1[r] * inv);
    s2[r] = (__bf16)(o2[r] * inv);
    s3[r] = (__bf16)(o3[r] * inv);
  }
  *(v8bf*)(out +  0) = s0;
  *(v8bf*)(out + 16) = s1;
  *(v8bf*)(out + 32) = s2;
  *(v8bf*)(out + 48) = s3;
}

// ---------------------------------------------------------------------------
// K3: out projection, 32x64 macro-tile per wave, fp32 output
// ---------------------------------------------------------------------------
__global__ __launch_bounds__(256) void out_gemm(const __bf16* __restrict__ X,
                                                const __bf16* __restrict__ W,
                                                const float*  __restrict__ bias,
                                                float* __restrict__ Y) {
  const int wave   = threadIdx.x >> 5;
  const int lane   = threadIdx.x & 31;
  const int waveId = blockIdx.x * 8 + wave;
  const int mtile  = waveId >> 4;
  const int ntile  = waveId & 15;
  const int laneN  = lane & 15;
  const int hi16   = lane >> 4;
  const int kbA = hi16 * 8;
  const int kbB = hi16 * 16;

  const __bf16* arow0 = X + (size_t)(mtile * 32 + laneN) * D_MODEL;
  const __bf16* arow1 = arow0 + (size_t)16 * D_MODEL;
  const __bf16* brow[4];
#pragma unroll
  for (int t = 0; t < 4; ++t)
    brow[t] = W + (size_t)(ntile * 64 + t * 16 + laneN) * D_MODEL;

  v8f acc[2][4];
#pragma unroll
  for (int mi = 0; mi < 2; ++mi)
#pragma unroll
    for (int t = 0; t < 4; ++t) acc[mi][t] = (v8f){};

#pragma unroll 2
  for (int k = 0; k < D_MODEL; k += 32) {
    v16bf a0 = frag_cat(arow0 + k + kbA, arow0 + k + kbA + 16);
    v16bf a1 = frag_cat(arow1 + k + kbA, arow1 + k + kbA + 16);
#pragma unroll
    for (int t = 0; t < 4; ++t) {
      v16bf b = frag_cat(brow[t] + k + kbA, brow[t] + k + kbA + 16);
      acc[0][t] = wmma_bf16(a0, b, acc[0][t]);
      acc[1][t] = wmma_bf16(a1, b, acc[1][t]);
    }
  }

  const int mbase = hi16 * 8;
#pragma unroll
  for (int mi = 0; mi < 2; ++mi) {
    const int rowb = mtile * 32 + mi * 16 + mbase;
#pragma unroll
    for (int t = 0; t < 4; ++t) {
      const int n = ntile * 64 + t * 16 + laneN;
      const float bv = bias[n];
#pragma unroll
      for (int r = 0; r < 8; ++r)
        Y[(size_t)(rowb + r) * D_MODEL + n] = acc[mi][t][r] + bv;
    }
  }
}

// ---------------------------------------------------------------------------
// host launcher
// ---------------------------------------------------------------------------
extern "C" void kernel_launch(void* const* d_in, const int* in_sizes, int n_in,
                              void* d_out, int out_size, void* d_ws, size_t ws_size,
                              hipStream_t stream) {
  const float* q  = (const float*)d_in[0];
  const float* k  = (const float*)d_in[1];
  const float* v  = (const float*)d_in[2];
  const float* Wq = (const float*)d_in[3];
  const float* bq = (const float*)d_in[4];
  const float* Wk = (const float*)d_in[5];
  const float* bk = (const float*)d_in[6];
  const float* Wv = (const float*)d_in[7];
  const float* bv = (const float*)d_in[8];
  const float* Wo = (const float*)d_in[9];
  const float* bo = (const float*)d_in[10];

  const size_t SZ_X = (size_t)ROWS * D_MODEL * sizeof(__bf16);      // 8 MiB
  const size_t SZ_W = (size_t)D_MODEL * D_MODEL * sizeof(__bf16);   // 2 MiB

  char* p = (char*)d_ws;
  __bf16* qbf = (__bf16*)p; p += SZ_X;
  __bf16* kbf = (__bf16*)p; p += SZ_X;
  __bf16* vbf = (__bf16*)p; p += SZ_X;
  __bf16* wqb = (__bf16*)p; p += SZ_W;
  __bf16* wkb = (__bf16*)p; p += SZ_W;
  __bf16* wvb = (__bf16*)p; p += SZ_W;
  __bf16* wob = (__bf16*)p; p += SZ_W;
  __bf16* Qh  = (__bf16*)p; p += SZ_X;
  __bf16* Kh  = (__bf16*)p; p += SZ_X;
  __bf16* Vt  = (__bf16*)p; p += SZ_X;   // transposed [b,h,d,s]
  __bf16* ctx = (__bf16*)p; p += SZ_X;

  const int NX = ROWS * D_MODEL;       // 4194304
  const int NW = D_MODEL * D_MODEL;    // 1048576

  // K0: downconvert activations + weights
  cvt_f32_bf16<<<2048, 256, 0, stream>>>(q,  qbf, NX);
  cvt_f32_bf16<<<2048, 256, 0, stream>>>(k,  kbf, NX);
  cvt_f32_bf16<<<2048, 256, 0, stream>>>(v,  vbf, NX);
  cvt_f32_bf16<<<512,  256, 0, stream>>>(Wq, wqb, NW);
  cvt_f32_bf16<<<512,  256, 0, stream>>>(Wk, wkb, NW);
  cvt_f32_bf16<<<512,  256, 0, stream>>>(Wv, wvb, NW);
  cvt_f32_bf16<<<512,  256, 0, stream>>>(Wo, wob, NW);

  // K1: QKV projections (2048 waves each, 32x64 tile per wave)
  proj_gemm_head<0><<<256, 256, 0, stream>>>(qbf, wqb, bq, Qh);
  proj_gemm_head<0><<<256, 256, 0, stream>>>(kbf, wkb, bk, Kh);
  proj_gemm_head<1><<<256, 256, 0, stream>>>(vbf, wvb, bv, Vt);

  // K2: flash attention (4096 waves, transposed pipeline, no LDS)
  attn_flash<<<512, 256, 0, stream>>>(Qh, Kh, Vt, ctx);

  // K3: output projection -> fp32 d_out
  out_gemm<<<256, 256, 0, stream>>>(ctx, wob, bo, (float*)d_out);
}
